// Block_65901978190163
// MI455X (gfx1250) — matrix operand order repaired and checked by reading, not compiled, not run on hardware
//
#include <hip/hip_runtime.h>
#include <hip/hip_bf16.h>
#include <stdint.h>

// ---------------------------------------------------------------------------
// GPT block forward on gfx1250 (MI455X): f16 WMMA compute, f32 accumulate.
//   ln1 -> qkv gemm -> flash attention -> proj+res -> ln2 -> up+gelu -> down+res
// GEMMs are A[M,K] * W[N,K]^T (K-major both sides) mapping directly onto the
// WMMA fragments in contiguous k order (k = 16h + s, A and B alike). GEMM waves compute 64x64 register
// tiles (16 wmma / K-step, ~32 FLOP/B from L2). Attention stages K tiles to
// LDS with GLOBAL_LOAD_ASYNC_TO_LDS_B128 (ASYNCcnt path) and reduces softmax
// rows with single-instruction ds_swizzle XOR butterflies.
// ---------------------------------------------------------------------------

typedef _Float16 v16h __attribute__((ext_vector_type(16)));
typedef _Float16 v8h  __attribute__((ext_vector_type(8)));
typedef float    v8f  __attribute__((ext_vector_type(8)));

#define N_EMBD 1024
#define N_HEAD 16
#define HEADD  64
#define BATCH  2
#define SEQ    2048
#define ROWS   (BATCH * SEQ)   // 4096

__device__ __forceinline__ v8f wmma32(v16h a, v16h b, v8f c) {
  // (neg_a, A, neg_b, B, c_mod, C, reuse_a, reuse_b)
  return __builtin_amdgcn_wmma_f32_16x16x32_f16(false, a, false, b, (short)0, c,
                                                false, false);
}

// XOR lane swap via ds_swizzle_b32 (group-of-32 mode: and=0x1f, or=0, xor=XM).
// Single LDS-permute instruction; masks <= 8 stay within each 16-lane half.
template <int XM>
__device__ __forceinline__ float swz_xor(float v) {
  return __int_as_float(
      __builtin_amdgcn_ds_swizzle(__float_as_int(v), (XM << 10) | 0x1f));
}
__device__ __forceinline__ float redmax16(float v) {
  v = fmaxf(v, swz_xor<1>(v));
  v = fmaxf(v, swz_xor<2>(v));
  v = fmaxf(v, swz_xor<4>(v));
  v = fmaxf(v, swz_xor<8>(v));
  return v;
}
__device__ __forceinline__ float redsum16(float v) {
  v += swz_xor<1>(v);
  v += swz_xor<2>(v);
  v += swz_xor<4>(v);
  v += swz_xor<8>(v);
  return v;
}

// A fragment 16x32 f16: lane<16 -> row, K k0..k0+15 contiguous;
// lane>=16 -> same rows, K k0+16..k0+31. Two contiguous 16B loads per lane.
__device__ __forceinline__ v16h load_a_frag(const _Float16* base, int row0,
                                            int ld, int k0) {
  int lane = threadIdx.x & 31;
  int lh   = lane >> 4;
  const _Float16* p0 = base + (size_t)(row0 + (lane & 15)) * ld + k0 + lh * 16;
  v8h a0 = *(const v8h*)p0;
  v8h a1 = *(const v8h*)(p0 + 8);
  v16h a;
#pragma unroll
  for (int i = 0; i < 8; ++i) { a[i] = a0[i]; a[8 + i] = a1[i]; }
  return a;
}

// B fragment 32x16 (K x N) from K-major weight rows: lane<16 holds row n0+lane,
// K k0..k0+15 contiguous; lane>=16 holds K k0+16..k0+31. One 32B load per lane.
__device__ __forceinline__ v16h load_b_frag(const _Float16* base, int n0,
                                            int ld, int k0) {
  int lane = threadIdx.x & 31;
  int lh   = lane >> 4;
  return *(const v16h*)(base + (size_t)(n0 + (lane & 15)) * ld + k0 + lh * 16);
}

// ---------------------------------------------------------------------------
__global__ void f32_to_f16(const float* __restrict__ in,
                           _Float16* __restrict__ out, size_t n) {
  size_t i = (size_t)blockIdx.x * blockDim.x + threadIdx.x;
  size_t stride = (size_t)gridDim.x * blockDim.x;
  for (; i < n; i += stride) out[i] = (_Float16)in[i];
}

// ---------------------------------------------------------------------------
// LayerNorm over 1024 features, one block (128 threads) per row, f16 output.
__global__ void ln_f16(const float* __restrict__ x, const float* __restrict__ w,
                       const float* __restrict__ b, _Float16* __restrict__ out) {
  int row = blockIdx.x;
  const float* xr = x + (size_t)row * N_EMBD;
  float v[8], s = 0.f, ss = 0.f;
#pragma unroll
  for (int i = 0; i < 8; ++i) {
    float t = xr[threadIdx.x + i * 128];
    v[i] = t; s += t; ss += t * t;
  }
  __shared__ float rs[128], rss[128];
  rs[threadIdx.x] = s; rss[threadIdx.x] = ss;
  __syncthreads();
  for (int off = 64; off > 0; off >>= 1) {
    if ((int)threadIdx.x < off) {
      rs[threadIdx.x]  += rs[threadIdx.x + off];
      rss[threadIdx.x] += rss[threadIdx.x + off];
    }
    __syncthreads();
  }
  float mean = rs[0] * (1.0f / N_EMBD);
  float var  = rss[0] * (1.0f / N_EMBD) - mean * mean;
  float rstd = rsqrtf(var + 1e-5f);
  _Float16* orow = out + (size_t)row * N_EMBD;
#pragma unroll
  for (int i = 0; i < 8; ++i) {
    int c = threadIdx.x + i * 128;
    orow[c] = (_Float16)((v[i] - mean) * rstd * w[c] + b[c]);
  }
}

// ---------------------------------------------------------------------------
// GEMM: out[M,N] = A[M,K] * W[N,K]^T + bias.
// One wave computes a 64x64 register tile: 4 A frags x 4 B frags -> 16 WMMAs
// per K step (each fragment reused 4x). 4 waves/block -> 64x256 per block.
// MODE 0: f16 out.  MODE 1: f16 out with exact GELU.  MODE 2: f32 out + resid.
template <int MODE>
__global__ void gemm_nt(const _Float16* __restrict__ A,
                        const _Float16* __restrict__ W,
                        const float* __restrict__ bias,
                        const float* __restrict__ resid,
                        float* __restrict__ outf, _Float16* __restrict__ outh,
                        int N, int K) {
  int wave = threadIdx.x >> 5;
  int lane = threadIdx.x & 31;
  int lh   = lane >> 4;
  int m0   = blockIdx.x * 64;
  int n0   = blockIdx.y * 256 + wave * 64;

  v8f acc[4][4] = {};
  for (int k = 0; k < K; k += 32) {
    if (k + 32 < K) {  // warm next chunks into cache (global_prefetch path)
      __builtin_prefetch(A + (size_t)(m0 + (lane & 15)) * K + k + 32, 0, 0);
      __builtin_prefetch(W + (size_t)(n0 + (lane & 15)) * K + k + 32, 0, 0);
    }
    v16h a[4];
#pragma unroll
    for (int i = 0; i < 4; ++i) a[i] = load_a_frag(A, m0 + i * 16, K, k);
#pragma unroll
    for (int j = 0; j < 4; ++j) {
      v16h b = load_b_frag(W, n0 + j * 16, K, k);
#pragma unroll
      for (int i = 0; i < 4; ++i) acc[i][j] = wmma32(a[i], b, acc[i][j]);
    }
  }
  // C/D layout: VGPR e -> row + e + 8*lh, col = lane&15 (ISA 7.12.2)
#pragma unroll
  for (int ti = 0; ti < 4; ++ti) {
#pragma unroll
    for (int j = 0; j < 4; ++j) {
#pragma unroll
      for (int e = 0; e < 8; ++e) {
        int row = m0 + ti * 16 + e + lh * 8;
        int n   = n0 + j * 16 + (lane & 15);
        float v = acc[ti][j][e] + bias[n];
        size_t idx = (size_t)row * N + n;
        if (MODE == 0) {
          outh[idx] = (_Float16)v;
        } else if (MODE == 1) {
          outh[idx] = (_Float16)(0.5f * v * (1.0f + erff(v * 0.70710678f)));
        } else {
          outf[idx] = resid[idx] + v;
        }
      }
    }
  }
}

// ---------------------------------------------------------------------------
// Causal flash attention. Block = one (b, head, 64-row q tile); 4 waves, each
// owning 16 q rows. K chunks staged to LDS via GLOBAL_LOAD_ASYNC_TO_LDS_B128
// (ASYNCcnt); V staged transposed so the att*V B fragment is a contiguous LDS
// read. Online softmax in f32 with ds_swizzle butterfly reductions.
__global__ void attn_kernel(const _Float16* __restrict__ qkv,
                            _Float16* __restrict__ out) {
  __shared__ __align__(32) _Float16 Kc[32 * 64];     // [kpos][d]
  __shared__ __align__(32) _Float16 Vt[64 * 32];     // [d][kpos]
  __shared__ __align__(32) _Float16 Pb[4][16 * 32];  // per-wave P tile

  const int tiles = SEQ / 64;
  int bh   = blockIdx.x / tiles;
  int qblk = blockIdx.x % tiles;
  int b    = bh / N_HEAD;
  int h    = bh % N_HEAD;
  int wave = threadIdx.x >> 5;
  int lane = threadIdx.x & 31;
  int lh   = lane >> 4;
  int qbase = qblk * 64;
  int q0    = qbase + wave * 16;

  const _Float16* base = qkv + (size_t)b * SEQ * (3 * N_EMBD) + h * HEADD;

  // Q fragments for this wave's 16 rows (d = 0..63 -> two K=32 fragments)
  v16h aq0 = load_a_frag(base, q0, 3 * N_EMBD, 0);
  v16h aq1 = load_a_frag(base, q0, 3 * N_EMBD, 32);

  v8f o[4] = {};
  float m[8], l[8];
#pragma unroll
  for (int i = 0; i < 8; ++i) { m[i] = -1e30f; l[i] = 0.f; }
  const float scale = 0.125f;  // 1/sqrt(64)

  for (int kc = 0; kc < qbase + 64; kc += 32) {
    __syncthreads();
    {  // cooperative stage: K via async copy (global -> LDS, no VGPR bounce)
      int r    = threadIdx.x >> 2;  // 0..31
      int part = threadIdx.x & 3;   // 0..3
      const _Float16* krow =
          base + (size_t)(kc + r) * (3 * N_EMBD) + N_EMBD + part * 16;
      // Flat LDS aperture keeps the LDS byte offset in addr[31:0] (ISA 10.2).
      unsigned ldsK = (unsigned)(uintptr_t)&Kc[r * 64 + part * 16];
      asm volatile(
          "global_load_async_to_lds_b128 %0, %1, off\n\t"
          "global_load_async_to_lds_b128 %2, %3, off"
          :
          : "v"(ldsK), "v"((unsigned long long)(uintptr_t)krow),
            "v"(ldsK + 16u), "v"((unsigned long long)(uintptr_t)(krow + 8))
          : "memory");
      // V staged transposed (manual: transpose is not an async-copy shape)
      const _Float16* vrow =
          base + (size_t)(kc + r) * (3 * N_EMBD) + 2 * N_EMBD + part * 16;
      v8h v0v = *(const v8h*)vrow;
      v8h v1v = *(const v8h*)(vrow + 8);
#pragma unroll
      for (int j = 0; j < 8; ++j) {
        Vt[(part * 16 + j) * 32 + r]     = v0v[j];
        Vt[(part * 16 + 8 + j) * 32 + r] = v1v[j];
      }
      asm volatile("s_wait_asynccnt 0x0" ::: "memory");
    }
    __syncthreads();
    if (kc > q0 + 15) continue;  // fully above the causal diagonal (wave-uniform)

    // S = q * K^T : two 16x16 subtiles over this 32-wide k chunk
    v8f s0 = {}, s1 = {};
    v16h bk;
    bk = *(const v16h*)&Kc[((lane & 15)) * 64 + lh * 16];            s0 = wmma32(aq0, bk, s0);
    bk = *(const v16h*)&Kc[((lane & 15)) * 64 + 32 + lh * 16];       s0 = wmma32(aq1, bk, s0);
    bk = *(const v16h*)&Kc[(16 + (lane & 15)) * 64 + lh * 16];       s1 = wmma32(aq0, bk, s1);
    bk = *(const v16h*)&Kc[(16 + (lane & 15)) * 64 + 32 + lh * 16];  s1 = wmma32(aq1, bk, s1);

    int col0 = kc + (lane & 15);
    int col1 = col0 + 16;
#pragma unroll
    for (int i = 0; i < 8; ++i) {
      int rowg = q0 + i + lh * 8;
      float a = (col0 <= rowg) ? s0[i] * scale : -1e30f;
      float c = (col1 <= rowg) ? s1[i] * scale : -1e30f;
      float mnew = fmaxf(m[i], redmax16(fmaxf(a, c)));
      float p0 = __expf(a - mnew);
      float p1 = __expf(c - mnew);
      float rsum = redsum16(p0 + p1);
      float corr = __expf(m[i] - mnew);
      l[i] = l[i] * corr + rsum;
      m[i] = mnew;
#pragma unroll
      for (int j = 0; j < 4; ++j) o[j][i] *= corr;
      _Float16* pr = &Pb[wave][(i + lh * 8) * 32 + (lane & 15)];
      pr[0]  = (_Float16)p0;
      pr[16] = (_Float16)p1;
    }

    // Reshape P (C layout) -> A fragment via per-wave LDS tile
    v16h ap;
    {
      const _Float16* pp = &Pb[wave][(lane & 15) * 32 + lh * 16];
      v8h x0 = *(const v8h*)pp;
      v8h x1 = *(const v8h*)(pp + 8);
#pragma unroll
      for (int i = 0; i < 8; ++i) { ap[i] = x0[i]; ap[8 + i] = x1[i]; }
    }
    // O += P * V ; Vt rows are d, so B fragment is a contiguous LDS read
#pragma unroll
    for (int j = 0; j < 4; ++j) {
      v16h bv = *(const v16h*)&Vt[(j * 16 + (lane & 15)) * 32 + lh * 16];
      o[j] = wmma32(ap, bv, o[j]);
    }
  }

  // Normalize and scatter to [B,T,C] f16
#pragma unroll
  for (int i = 0; i < 8; ++i) {
    float invl = 1.0f / l[i];
    int t = q0 + i + lh * 8;
    _Float16* orow =
        out + (size_t)(b * SEQ + t) * N_EMBD + h * HEADD + (lane & 15);
#pragma unroll
    for (int j = 0; j < 4; ++j) orow[j * 16] = (_Float16)(o[j][i] * invl);
  }
}

// ---------------------------------------------------------------------------
extern "C" void kernel_launch(void* const* d_in, const int* in_sizes, int n_in,
                              void* d_out, int out_size, void* d_ws,
                              size_t ws_size, hipStream_t stream) {
  (void)in_sizes; (void)n_in; (void)out_size; (void)ws_size;
  const float* x      = (const float*)d_in[0];
  const float* ln1w   = (const float*)d_in[1];
  const float* ln1b   = (const float*)d_in[2];
  const float* attn_w = (const float*)d_in[3];
  const float* attn_b = (const float*)d_in[4];
  const float* proj_w = (const float*)d_in[5];
  const float* proj_b = (const float*)d_in[6];
  const float* ln2w   = (const float*)d_in[7];
  const float* ln2b   = (const float*)d_in[8];
  const float* up_w   = (const float*)d_in[9];
  const float* up_b   = (const float*)d_in[10];
  const float* down_w = (const float*)d_in[11];
  const float* down_b = (const float*)d_in[12];
  float* out = (float*)d_out;

  _Float16* p = (_Float16*)d_ws;
  _Float16* wAttn = p; p += (size_t)3 * N_EMBD * N_EMBD;
  _Float16* wProj = p; p += (size_t)N_EMBD * N_EMBD;
  _Float16* wUp   = p; p += (size_t)4 * N_EMBD * N_EMBD;
  _Float16* wDown = p; p += (size_t)4 * N_EMBD * N_EMBD;
  _Float16* h16   = p; p += (size_t)ROWS * N_EMBD;
  _Float16* qkv16 = p; p += (size_t)ROWS * 3 * N_EMBD;
  _Float16* att16 = p; p += (size_t)ROWS * N_EMBD;
  _Float16* up16  = p; p += (size_t)ROWS * 4 * N_EMBD;

  auto cvt = [&](const float* src, _Float16* dst, size_t n) {
    f32_to_f16<<<dim3((unsigned)((n + 1023) / 1024)), dim3(256), 0, stream>>>(
        src, dst, n);
  };
  cvt(attn_w, wAttn, (size_t)3 * N_EMBD * N_EMBD);
  cvt(proj_w, wProj, (size_t)N_EMBD * N_EMBD);
  cvt(up_w,   wUp,   (size_t)4 * N_EMBD * N_EMBD);
  cvt(down_w, wDown, (size_t)4 * N_EMBD * N_EMBD);

  // ln1(x) -> h16
  ln_f16<<<dim3(ROWS), dim3(128), 0, stream>>>(x, ln1w, ln1b, h16);
  // qkv = h16 * Wqkv^T + b  (f16 out)
  gemm_nt<0><<<dim3(ROWS / 64, (3 * N_EMBD) / 256), dim3(128), 0, stream>>>(
      h16, wAttn, attn_b, nullptr, nullptr, qkv16, 3 * N_EMBD, N_EMBD);
  // causal attention -> att16
  attn_kernel<<<dim3(BATCH * N_HEAD * (SEQ / 64)), dim3(128), 0, stream>>>(
      qkv16, att16);
  // out = x + att16 * Wproj^T + b  (f32)
  gemm_nt<2><<<dim3(ROWS / 64, N_EMBD / 256), dim3(128), 0, stream>>>(
      att16, wProj, proj_b, x, out, nullptr, N_EMBD, N_EMBD);
  // ln2(out) -> h16
  ln_f16<<<dim3(ROWS), dim3(128), 0, stream>>>(out, ln2w, ln2b, h16);
  // up16 = gelu(h16 * Wup^T + b)  (f16)
  gemm_nt<1><<<dim3(ROWS / 64, (4 * N_EMBD) / 256), dim3(128), 0, stream>>>(
      h16, wUp, up_b, nullptr, nullptr, up16, 4 * N_EMBD, N_EMBD);
  // out = out + up16 * Wdown^T + b  (f32, per-tile read-then-write, no races)
  gemm_nt<2><<<dim3(ROWS / 64, N_EMBD / 256), dim3(128), 0, stream>>>(
      up16, wDown, down_b, out, out, nullptr, N_EMBD, 4 * N_EMBD);
}
